// ColumnEncoding_33320356282773
// MI455X (gfx1250) — matrix operand order repaired and checked by reading, not compiled, not run on hardware
//
#include <hip/hip_runtime.h>
#include <hip/hip_bf16.h>
#include <stdint.h>
#include <stddef.h>

// ---------------- problem constants (match reference) ----------------
#define VOCAB   32000
#define EMBED   300
#define HID     256
#define NCOLS   4096
#define SEGLEN  8
#define GATES   1024          // 4*HID
#define KEMB    320           // EMBED padded to multiple of 32 (zeros in [300,320))
#define KTOT    576           // KEMB + HID  (18 * 32)
#define ASTR    584           // LDS row stride (bf16 elems): 292 dwords -> bank-conflict free

typedef __attribute__((ext_vector_type(16))) __bf16 v16bf;
typedef __attribute__((ext_vector_type(8)))  float  v8f;

struct __align__(16) U4 { unsigned x, y, z, w; };
union Frag { v16bf v; U4 q[2]; };

__device__ __forceinline__ float fsig(float x) {
  // 1/(1+e^-x) via v_exp_f32 + v_rcp_f32 (trans pipe, co-executes with WMMA)
  return __builtin_amdgcn_rcpf(1.0f + __expf(-x));
}
__device__ __forceinline__ float ftanh(float x) {
  // tanh(x) = 1 - 2/(e^{2x}+1); saturates correctly at +/-inf
  float e = __expf(2.0f * x);
  return 1.0f - 2.0f * __builtin_amdgcn_rcpf(e + 1.0f);
}

// ------------------------------------------------------------------
// Prep 1: pack [W_ih | 0-pad | W_hh] per direction into bf16 [2][1024][576],
// and combined bias b_ih+b_hh into fp32 [2][1024].
// ------------------------------------------------------------------
__global__ __launch_bounds__(256) void prep_w(
    const float* __restrict__ wih_f, const float* __restrict__ whh_f,
    const float* __restrict__ bih_f, const float* __restrict__ bhh_f,
    const float* __restrict__ wih_b, const float* __restrict__ whh_b,
    const float* __restrict__ bih_b, const float* __restrict__ bhh_b,
    __bf16* __restrict__ W, float* __restrict__ biasc)
{
  int idx = blockIdx.x * 256 + threadIdx.x;
  if (idx >= 2 * GATES * KTOT) return;
  int dir = idx / (GATES * KTOT);
  int rem = idx - dir * (GATES * KTOT);
  int g   = rem / KTOT;
  int k   = rem - g * KTOT;
  const float* wih = dir ? wih_b : wih_f;
  const float* whh = dir ? whh_b : whh_f;
  float v = 0.0f;
  if (k < EMBED)           v = wih[(size_t)g * EMBED + k];
  else if (k >= KEMB)      v = whh[(size_t)g * HID + (k - KEMB)];
  W[idx] = (__bf16)v;
  if (k == 0) {
    const float* bih = dir ? bih_b : bih_f;
    const float* bhh = dir ? bhh_b : bhh_f;
    biasc[dir * GATES + g] = bih[g] + bhh[g];
  }
}

// ------------------------------------------------------------------
// Prep 2: convert the embedding table to bf16 with rows padded to 320 elems
// (zeros in [300,320)). Makes the per-step A staging a raw byte copy that
// the async-to-LDS path can move directly, and halves gather HBM traffic.
// ------------------------------------------------------------------
__global__ __launch_bounds__(256) void prep_emb(
    const float* __restrict__ emb, __bf16* __restrict__ embbf)
{
  int idx = blockIdx.x * 256 + threadIdx.x;
  if (idx >= VOCAB * KEMB) return;
  int row = idx / KEMB;
  int k   = idx - row * KEMB;
  float v = (k < EMBED) ? emb[(size_t)row * EMBED + k] : 0.0f;
  embbf[idx] = (__bf16)v;
}

// Zero the recurrent state (must happen every call: harness replays the graph).
__global__ __launch_bounds__(256) void init_state(
    float* __restrict__ Cst, float* __restrict__ Hst, __bf16* __restrict__ Hbf)
{
  int idx = blockIdx.x * 256 + threadIdx.x;
  if (idx >= 2 * NCOLS * HID) return;
  Cst[idx] = 0.0f;
  Hst[idx] = 0.0f;
  Hbf[idx] = (__bf16)0.0f;
}

// ------------------------------------------------------------------
// One LSTM timestep, both directions (blockIdx.z = dir), fully fused:
//   gates[n,g] = [emb(tok(n,l)) | h_prev(n)] @ [W_ih | W_hh]^T + (b_ih+b_hh)
// GEMM: M=16 rows(n) per WG, N=1024 gate cols, K=576, bf16 WMMA f32-acc.
// A tile staged with GLOBAL_LOAD_ASYNC_TO_LDS_B128 (ASYNCcnt path, no VGPR
// round trip). Wave w owns j in [w*32, w*32+32) across ALL four gates
// (8 C-tiles), so the cell update needs no cross-lane traffic.
// ------------------------------------------------------------------
__global__ __launch_bounds__(256) void lstm_step(
    const __bf16* __restrict__ Ebf, const int* __restrict__ seq,
    const __bf16* __restrict__ W, const float* __restrict__ biasc,
    float* __restrict__ Cst, float* __restrict__ Hst, __bf16* __restrict__ Hbf,
    int step)
{
  __shared__ __bf16 Alds[16 * ASTR];           // 18.25 KB

  const int dir = blockIdx.z;
  const int n0  = blockIdx.x * 16;
  const int l   = dir ? (SEGLEN - 1 - step) : step;   // backward runs reversed time
  const int tid = threadIdx.x;

  // ---- async-stage A tile: 16 rows x [embbf 320 | h_prev bf16 256] ----
  // Row = 40 chunks (emb, 640B) + 32 chunks (h_prev, 512B) of 16B each.
  const int NCHUNK = 16 * 72;
  for (int c = tid; c < NCHUNK; c += 256) {
    int row = c / 72;
    int q   = c - row * 72;
    int n   = n0 + row;
    uint64_t gaddr;
    if (q < 40) {
      int t   = (n == 0 ? 0 : n * SEGLEN - 1) + l;    // begins[n] + l
      int tok = seq[t];
      gaddr = (uint64_t)(uintptr_t)(Ebf + (size_t)tok * KEMB + q * 8);
    } else {
      gaddr = (uint64_t)(uintptr_t)(Hbf + ((size_t)dir * NCOLS + n) * HID + (q - 40) * 8);
    }
    unsigned laddr = (unsigned)(uintptr_t)(&Alds[row * ASTR + q * 8]);
    asm volatile("global_load_async_to_lds_b128 %0, %1, off"
                 :: "v"(laddr), "v"(gaddr) : "memory");
  }
  asm volatile("s_wait_asynccnt 0x0" ::: "memory");
  __syncthreads();

  const int lane = tid & 31;
  const int wave = tid >> 5;
  const int m16  = lane & 15;
  const int hi   = lane >> 4;

  const v8f vzero = {0.f,0.f,0.f,0.f,0.f,0.f,0.f,0.f};
  v8f acc[4][2];
  #pragma unroll
  for (int g = 0; g < 4; ++g)
    #pragma unroll
    for (int t = 0; t < 2; ++t) acc[g][t] = vzero;

  const __bf16* Wd = W + (size_t)dir * GATES * KTOT;

  // ---- K loop: 18 iterations of 8 wmma_f32_16x16x32_bf16 ----
  for (int k0 = 0; k0 < KTOT; k0 += 32) {
    // A fragment (ISA 16-bit A 16x32 layout): lanes 0-15 K={k0..k0+7, k0+16..k0+23},
    // lanes 16-31 K shifted by 8 -> two contiguous 16B LDS loads.
    Frag a;
    const __bf16* ap = &Alds[m16 * ASTR + k0 + hi * 8];
    a.q[0] = *reinterpret_cast<const U4*>(ap);
    a.q[1] = *reinterpret_cast<const U4*>(ap + 16);
    #pragma unroll
    for (int g = 0; g < 4; ++g) {
      #pragma unroll
      for (int t = 0; t < 2; ++t) {
        // B fragment (32x16): lane col = gate*256 + wave*32 + t*16 + m16,
        // K={k0+hi*8..+7, k0+16+hi*8..+7} (matches A) -> two 16B global loads (L2-resident W).
        int gcol = g * HID + wave * 32 + t * 16 + m16;
        const __bf16* bp = Wd + (size_t)gcol * KTOT + k0 + hi * 8;
        Frag b;
        b.q[0] = *reinterpret_cast<const U4*>(bp);
        b.q[1] = *reinterpret_cast<const U4*>(bp + 16);
        acc[g][t] = __builtin_amdgcn_wmma_f32_16x16x32_bf16(
            false, a.v, false, b.v, (short)0, acc[g][t], false, false);
      }
    }
  }

  // ---- fused LSTM cell epilogue ----
  // C/D layout: VGPR v, lanes 0-15 -> M=v, lanes 16-31 -> M=v+8; N = lane%16.
  #pragma unroll
  for (int t = 0; t < 2; ++t) {
    int j = wave * 32 + t * 16 + m16;
    const float* bb = biasc + dir * GATES;
    float bi  = bb[0 * HID + j];
    float bff = bb[1 * HID + j];
    float bg  = bb[2 * HID + j];
    float bo  = bb[3 * HID + j];
    #pragma unroll
    for (int v = 0; v < 8; ++v) {
      int n   = n0 + v + hi * 8;
      int len = (n == 0) ? (SEGLEN - 1) : SEGLEN;
      size_t sidx = ((size_t)dir * NCOLS + n) * HID + j;
      float iv = fsig (acc[0][t][v] + bi);
      float fv = fsig (acc[1][t][v] + bff);
      float gv = ftanh(acc[2][t][v] + bg);
      float ov = fsig (acc[3][t][v] + bo);
      float cold = Cst[sidx];
      float cn = fv * cold + iv * gv;
      float hn = ov * ftanh(cn);
      if (l < len) {                         // masked update (only n==0, l==7 is masked)
        Cst[sidx] = cn;
        Hst[sidx] = hn;
        Hbf[sidx] = (__bf16)hn;              // bf16 copy feeds next step's A tile
      }
    }
  }
}

// out[n][0:256] = h_f[n], out[n][256:512] = h_b[n]   (fp32)
__global__ __launch_bounds__(256) void write_out(
    const float* __restrict__ Hst, float* __restrict__ out)
{
  int idx = blockIdx.x * 256 + threadIdx.x;
  if (idx >= NCOLS * 2 * HID) return;
  int n   = idx >> 9;          // /512
  int j   = idx & 511;
  int dir = j >> 8;
  int jj  = j & 255;
  out[idx] = Hst[((size_t)dir * NCOLS + n) * HID + jj];
}

// ------------------------------------------------------------------
extern "C" void kernel_launch(void* const* d_in, const int* in_sizes, int n_in,
                              void* d_out, int out_size, void* d_ws, size_t ws_size,
                              hipStream_t stream)
{
  const float* emb   = (const float*)d_in[0];
  const float* wih_f = (const float*)d_in[1];
  const float* whh_f = (const float*)d_in[2];
  const float* bih_f = (const float*)d_in[3];
  const float* bhh_f = (const float*)d_in[4];
  const float* wih_b = (const float*)d_in[5];
  const float* whh_b = (const float*)d_in[6];
  const float* bih_b = (const float*)d_in[7];
  const float* bhh_b = (const float*)d_in[8];
  const int*   seq   = (const int*)d_in[9];

  // workspace layout (~42 MB, 256B-aligned slabs)
  char* ws = (char*)d_ws;
  __bf16* W     = (__bf16*)(ws + 0);            // 2*1024*576*2  =  2,359,296
  float*  biasc = (float*) (ws + 2359296);      // 2048*4        =      8,192
  float*  Cst   = (float*) (ws + 2367488);      // 2*4096*256*4  =  8,388,608
  float*  Hst   = (float*) (ws + 10756096);     //               =  8,388,608
  __bf16* Hbf   = (__bf16*)(ws + 19144704);     // 2*4096*256*2  =  4,194,304
  __bf16* Ebf   = (__bf16*)(ws + 23339008);     // 32000*320*2   = 20,480,000

  prep_w<<<dim3((2 * GATES * KTOT + 255) / 256), 256, 0, stream>>>(
      wih_f, whh_f, bih_f, bhh_f, wih_b, whh_b, bih_b, bhh_b, W, biasc);
  prep_emb<<<dim3((VOCAB * KEMB + 255) / 256), 256, 0, stream>>>(emb, Ebf);
  init_state<<<dim3((2 * NCOLS * HID + 255) / 256), 256, 0, stream>>>(Cst, Hst, Hbf);

  for (int s = 0; s < SEGLEN; ++s)
    lstm_step<<<dim3(NCOLS / 16, 1, 2), 256, 0, stream>>>(
        Ebf, seq, W, biasc, Cst, Hst, Hbf, s);

  write_out<<<dim3((NCOLS * 2 * HID + 255) / 256), 256, 0, stream>>>(Hst, (float*)d_out);
}
